// QSAR_57810259804592
// MI455X (gfx1250) — matrix operand order repaired and checked by reading, not compiled, not run on hardware
//
#include <hip/hip_runtime.h>
#include <hip/hip_bf16.h>

// ---- CDNA5 WMMA types ----
typedef __attribute__((ext_vector_type(16))) __bf16          v16bf;
typedef __attribute__((ext_vector_type(16))) unsigned short  v16u;
typedef __attribute__((ext_vector_type(8)))  unsigned short  v8u;
typedef __attribute__((ext_vector_type(8)))  float           v8f;
typedef __attribute__((ext_vector_type(4)))  int             v4i;

#define A_   64
#define D_   6
#define AF_  37
#define BF_  6
#define H_   128
#define KPS_ 168   // LDS feature stride in halves (336B: conflict-free across 16 rows)

__device__ __forceinline__ unsigned short f2bf(float f) {
  // round-to-nearest-even f32 -> bf16
  unsigned u = __builtin_bit_cast(unsigned, f);
  u += 0x7FFFu + ((u >> 16) & 1u);
  return (unsigned short)(u >> 16);
}

// ---------------------------------------------------------------------------
// Weight prepack: f32 (ndeg, Kreal, 128) -> bf16 tiles in WMMA B-fragment order.
// Tile (d, kt, nt) is 512 halves; within a tile, lane L owns 16 contiguous halves:
//   col n = nt*16 + (L&15), k = kt*32 + 16*(L>>4) + i   (zero beyond Kreal)
// ---------------------------------------------------------------------------
__global__ void prepack_w(const float* __restrict__ W, unsigned short* __restrict__ out,
                          int Kreal, int Ktiles, int ndeg) {
  const int total = ndeg * Ktiles * 8 * 512;
  for (int i = blockIdx.x * blockDim.x + threadIdx.x; i < total;
       i += gridDim.x * blockDim.x) {
    const int e    = i & 511;
    const int tile = i >> 9;
    const int nt   = tile & 7;
    const int kt   = (tile >> 3) % Ktiles;
    const int d    = tile / (8 * Ktiles);
    const int lane = e >> 4;
    const int ii   = e & 15;
    const int n    = nt * 16 + (lane & 15);
    const int k    = kt * 32 + 16 * (lane >> 4) + ii;
    const float v  = (k < Kreal) ? W[((size_t)d * Kreal + k) * H_ + n] : 0.f;
    out[i] = f2bf(v);
  }
}

// ---------------------------------------------------------------------------
// 16x16 MAC tile: KT chained v_wmma_f32_16x16x32_bf16, returns C fragment.
// ---------------------------------------------------------------------------
template <int KT>
__device__ __forceinline__ v8f wmma_mm(
    const unsigned short* __restrict__ sF,
    const unsigned short* __restrict__ Wt,   // prepacked slice: KT*8 tiles of 512 halves
    int m0, int nt, int col, int kh, int lane)
{
  v8f c = {};
#pragma unroll
  for (int kt = 0; kt < KT; ++kt) {
    const int k0 = kt * 32;
    // A fragment: lane holds row (m0+col); halves {k0+16kh..+15}
    const v8u* ap = (const v8u*)(sF + (m0 + col) * KPS_ + k0 + 16 * kh);
    const v8u lo = ap[0];
    const v8u hi = ap[1];            // +8 halves
    v16u au;
#pragma unroll
    for (int i = 0; i < 8; ++i) { au[i] = lo[i]; au[8 + i] = hi[i]; }
    // B fragment: one contiguous 32B vector per lane (coalesced, unguarded)
    const v16u bu = *(const v16u*)(Wt + (((size_t)kt * 8 + nt) << 9) + (lane << 4));
    const v16bf a = __builtin_bit_cast(v16bf, au);
    const v16bf b = __builtin_bit_cast(v16bf, bu);
    c = __builtin_amdgcn_wmma_f32_16x16x32_bf16(false, a, false, b, (short)0, c,
                                                false, false);
  }
  return c;
}

// Degree-dispatched graph-conv GEMM: out[a,:] = relu(feats[a,:] @ W[deg[a]] + b[deg[a]])
// Branchless accumulation: per-lane outv[8] selected by v_cndmask, single store pass.
template <int KT>
__device__ __forceinline__ void conv_gemm(
    const unsigned short* __restrict__ sF, const unsigned short* __restrict__ Wpk,
    const float* __restrict__ sBias,         // 6*128 floats in LDS
    float* __restrict__ sOut, const int* __restrict__ sDeg, int tid)
{
  const int lane = tid & 31;
  const int nt   = tid >> 5;                 // 8 waves -> 8 N-tiles
  const int col  = lane & 15;
  const int kh   = lane >> 4;
  for (int m = 0; m < 4; ++m) {
    const int m0 = m * 16;
    // This lane's 8 output rows and their degrees (contiguous -> ds_load_b128 x2)
    const v4i* dp = (const v4i*)(sDeg + m0 + 8 * kh);
    const v4i d0 = dp[0], d1 = dp[1];
    int degv[8];
#pragma unroll
    for (int i = 0; i < 4; ++i) { degv[i] = d0[i]; degv[4 + i] = d1[i]; }
    unsigned dm = 0;
#pragma unroll
    for (int j = 0; j < 8; ++j) dm |= 1u << degv[j];
    // combine both lane-halves' row sets, force scalar
    const unsigned dmAll =
        (unsigned)__builtin_amdgcn_readfirstlane((int)dm) |
        (unsigned)__builtin_amdgcn_readlane((int)dm, 16);
    float outv[8];
#pragma unroll
    for (int j = 0; j < 8; ++j) outv[j] = 0.f;
    for (int d = 0; d < 6; ++d) {
      if (!((dmAll >> d) & 1u)) continue;    // uniform skip: EXEC stays all-1s
      const v8f c = wmma_mm<KT>(sF, Wpk + (size_t)d * KT * 8 * 512, m0, nt, col, kh, lane);
      const float bb = sBias[d * H_ + nt * 16 + col];
#pragma unroll
      for (int j = 0; j < 8; ++j) {
        const float v = fmaxf(c[j] + bb, 0.f);
        outv[j] = (degv[j] == d) ? v : outv[j];   // v_cndmask, no branch
      }
    }
#pragma unroll
    for (int j = 0; j < 8; ++j)
      sOut[(m0 + j + 8 * kh) * H_ + nt * 16 + col] = outv[j];
  }
}

__global__ __launch_bounds__(256)
void qsar_fused(const float* __restrict__ atoms, const float* __restrict__ bonds,
                const int* __restrict__ edges,
                const unsigned short* __restrict__ w1pk, const float* __restrict__ b1,
                const unsigned short* __restrict__ w2pk, const float* __restrict__ b2,
                const unsigned short* __restrict__ wopk, const float* __restrict__ bo,
                const float* __restrict__ Wm1, const float* __restrict__ bm1,
                const float* __restrict__ Wm2, const float* __restrict__ bm2,
                const float* __restrict__ Wm3, const float* __restrict__ bm3,
                float* __restrict__ out)
{
  __shared__ __align__(32) unsigned short sF[A_ * KPS_];  // bf16 feats      21.0 KB
  __shared__ float sX[A_ * H_];                           //                 32.0 KB
  __shared__ float sP[A_ * H_];                           //                 32.0 KB
  __shared__ float sAtoms[A_ * 40];                       //                 10.0 KB
  __shared__ float sBias[6 * H_ * 2 + H_];                // b1|b2|bo         6.5 KB
  __shared__ float sBsum[A_ * 8];
  __shared__ __align__(32) int sDeg[A_];
  __shared__ int   sEdge[A_ * 8];
  __shared__ float sFp[H_];
  __shared__ float sH1[64];
  __shared__ float sRed[32];
  __shared__ float sRR[256];                              // reductions

  const int b   = blockIdx.x;
  const int tid = threadIdx.x;
  const float* atomsB = atoms + (size_t)b * A_ * AF_;
  const float* bondsB = bonds + (size_t)b * A_ * D_ * BF_;
  const int*   edgesB = edges + (size_t)b * A_ * D_;

  // ---- Phase 0: stage edges/deg, atoms, bond sums, biases ----
  if (tid < A_) {
    int dg = 0;
    for (int j = 0; j < D_; ++j) {
      const int e = edgesB[tid * D_ + j];
      sEdge[tid * 8 + j] = e;
      if (e >= 0) dg++;
    }
    sDeg[tid] = dg;
  }
  for (int i = tid; i < A_ * AF_; i += 256)
    sAtoms[(i / AF_) * 40 + (i % AF_)] = atomsB[i];
  for (int i = tid; i < A_ * BF_; i += 256) {
    const int a = i / BF_, f = i % BF_;
    float s = 0.f;
    for (int j = 0; j < D_; ++j) s += bondsB[(a * D_ + j) * BF_ + f];
    sBsum[a * 8 + f] = s;
  }
  for (int i = tid; i < 6 * H_; i += 256) { sBias[i] = b1[i]; sBias[6 * H_ + i] = b2[i]; }
  if (tid < H_) sBias[12 * H_ + tid] = bo[tid];
  __syncthreads();

  // ---- feats1 = [self+neighbor atom sum (37) | bond sum (6) | zero pad -> 64] ----
  for (int i = tid; i < A_ * 64; i += 256) {
    const int a = i >> 6, k = i & 63;
    float v = 0.f;
    if (k < AF_) {
      v = sAtoms[a * 40 + k];
      for (int j = 0; j < D_; ++j) {
        const int e = sEdge[a * 8 + j];
        if (e >= 0) v += sAtoms[e * 40 + k];
      }
    } else if (k < AF_ + BF_) {
      v = sBsum[a * 8 + (k - AF_)];
    }
    sF[a * KPS_ + k] = f2bf(v);
  }
  __syncthreads();

  // ---- conv1: (64x43)@(43x128) per degree (2 WMMA K-steps) ----
  conv_gemm<2>(sF, w1pk, sBias, sX, sDeg, tid);
  __syncthreads();

  // ---- pool1: max over self+neighbors ----
  for (int i = tid; i < A_ * H_; i += 256) {
    const int a = i >> 7, c = i & (H_ - 1);
    float m = sX[i];
    for (int j = 0; j < D_; ++j) {
      const int e = sEdge[a * 8 + j];
      if (e >= 0) { const float t = sX[e * H_ + c]; m = t > m ? t : m; }
    }
    sP[i] = m;
  }
  __syncthreads();

  // ---- feats2 = [self+neighbor pooled sum (128) | bond sum (6) | pad -> 160] ----
  for (int i = tid; i < A_ * H_; i += 256) {
    const int a = i >> 7, k = i & (H_ - 1);
    float v = sP[i];
    for (int j = 0; j < D_; ++j) {
      const int e = sEdge[a * 8 + j];
      if (e >= 0) v += sP[e * H_ + k];
    }
    sF[a * KPS_ + k] = f2bf(v);
  }
  for (int i = tid; i < A_ * 32; i += 256) {
    const int a = i >> 5, k = H_ + (i & 31);
    sF[a * KPS_ + k] = (k < H_ + BF_) ? f2bf(sBsum[a * 8 + (k - H_)]) : (unsigned short)0;
  }
  __syncthreads();

  // ---- conv2: (64x134)@(134x128) per degree (5 WMMA K-steps) ----
  conv_gemm<5>(sF, w2pk, sBias + 6 * H_, sX, sDeg, tid);
  __syncthreads();

  // ---- pool2 ----
  for (int i = tid; i < A_ * H_; i += 256) {
    const int a = i >> 7, c = i & (H_ - 1);
    float m = sX[i];
    for (int j = 0; j < D_; ++j) {
      const int e = sEdge[a * 8 + j];
      if (e >= 0) { const float t = sX[e * H_ + c]; m = t > m ? t : m; }
    }
    sP[i] = m;
  }
  __syncthreads();

  // ---- feats3 = [pooled (128, no neighbor sum) | bond sum | pad] ----
  for (int i = tid; i < A_ * H_; i += 256) {
    const int a = i >> 7, k = i & (H_ - 1);
    sF[a * KPS_ + k] = f2bf(sP[i]);
  }
  for (int i = tid; i < A_ * 32; i += 256) {
    const int a = i >> 5, k = H_ + (i & 31);
    sF[a * KPS_ + k] = (k < H_ + BF_) ? f2bf(sBsum[a * 8 + (k - H_)]) : (unsigned short)0;
  }
  __syncthreads();

  // ---- output GEMM: logits = feats3 @ Wo + bo (all rows, unconditional) ----
  {
    const int lane = tid & 31;
    const int nt   = tid >> 5;
    const int col  = lane & 15;
    const int kh   = lane >> 4;
    for (int m = 0; m < 4; ++m) {
      const v8f c = wmma_mm<5>(sF, wopk, m * 16, nt, col, kh, lane);
      const float bb = sBias[12 * H_ + nt * 16 + col];
#pragma unroll
      for (int j = 0; j < 8; ++j)
        sX[(m * 16 + j + 8 * kh) * H_ + nt * 16 + col] = c[j] + bb;
    }
  }
  __syncthreads();

  // ---- per-atom softmax (4 threads/row), zero deg==0 rows ----
  {
    const int a = tid >> 2, c0 = (tid & 3) * 32;
    float mx = -3.0e38f;
    for (int c = c0; c < c0 + 32; ++c) mx = fmaxf(mx, sX[a * H_ + c]);
    sRR[tid] = mx;
    __syncthreads();
    mx = fmaxf(fmaxf(sRR[a * 4 + 0], sRR[a * 4 + 1]),
               fmaxf(sRR[a * 4 + 2], sRR[a * 4 + 3]));
    float s = 0.f;
    for (int c = c0; c < c0 + 32; ++c) {
      const float e = __expf(sX[a * H_ + c] - mx);
      sX[a * H_ + c] = e;
      s += e;
    }
    __syncthreads();
    sRR[tid] = s;
    __syncthreads();
    const float tot = sRR[a * 4 + 0] + sRR[a * 4 + 1] + sRR[a * 4 + 2] + sRR[a * 4 + 3];
    const float scale = (sDeg[a] == 0) ? 0.f : 1.f / tot;
    for (int c = c0; c < c0 + 32; ++c) sX[a * H_ + c] *= scale;
  }
  __syncthreads();

  // ---- molecular fingerprint: sum over atoms (2 threads/col) ----
  {
    const int c = tid & 127, half = tid >> 7;
    float s = 0.f;
    for (int a = half * 32; a < half * 32 + 32; ++a) s += sX[a * H_ + c];
    sRR[tid] = s;
  }
  __syncthreads();
  if (tid < H_) sFp[tid] = sRR[tid] + sRR[128 + tid];
  __syncthreads();

  // ---- MLP head: 128 -> 64 -> 32 -> 1 ----
  if (tid < 64) {
    float s = bm1[tid];
    for (int k = 0; k < H_; ++k) s += sFp[k] * Wm1[k * 64 + tid];
    sH1[tid] = s > 0.f ? s : 0.f;
  }
  __syncthreads();
  if (tid < 32) {
    float s = bm2[tid];
    for (int k = 0; k < 64; ++k) s += sH1[k] * Wm2[k * 32 + tid];
    s = s > 0.f ? s : 0.f;
    sRed[tid] = s * Wm3[tid];
  }
  __syncthreads();
  if (tid == 0) {
    float s = bm3[0];
    for (int k = 0; k < 32; ++k) s += sRed[k];
    out[b] = s;
  }
}

extern "C" void kernel_launch(void* const* d_in, const int* in_sizes, int n_in,
                              void* d_out, int out_size, void* d_ws, size_t ws_size,
                              hipStream_t stream) {
  (void)n_in; (void)ws_size; (void)out_size;
  const float* atoms = (const float*)d_in[0];
  const float* bonds = (const float*)d_in[1];
  const int*   edges = (const int*)d_in[2];
  const float* W1  = (const float*)d_in[3];
  const float* b1  = (const float*)d_in[4];
  const float* W2  = (const float*)d_in[5];
  const float* b2  = (const float*)d_in[6];
  const float* Wo  = (const float*)d_in[7];
  const float* bo  = (const float*)d_in[8];
  const float* Wm1 = (const float*)d_in[9];
  const float* bm1 = (const float*)d_in[10];
  const float* Wm2 = (const float*)d_in[11];
  const float* bm2 = (const float*)d_in[12];
  const float* Wm3 = (const float*)d_in[13];
  const float* bm3 = (const float*)d_in[14];
  float* out = (float*)d_out;

  // Prepacked bf16 weight fragments in workspace (~385 KB total, L2-resident)
  unsigned short* w1pk = (unsigned short*)d_ws;          // 6*2*8*512 halves
  unsigned short* w2pk = w1pk + 6 * 2 * 8 * 512;         // 6*5*8*512 halves
  unsigned short* wopk = w2pk + 6 * 5 * 8 * 512;         // 1*5*8*512 halves

  prepack_w<<<64, 256, 0, stream>>>(W1, w1pk, AF_ + BF_, 2, 6);
  prepack_w<<<128, 256, 0, stream>>>(W2, w2pk, H_ + BF_, 5, 6);
  prepack_w<<<32, 256, 0, stream>>>(Wo, wopk, H_ + BF_, 5, 1);

  const int B = in_sizes[0] / (A_ * AF_);   // 4096 molecules
  qsar_fused<<<B, 256, 0, stream>>>(atoms, bonds, edges, w1pk, b1, w2pk, b2,
                                    wopk, bo, Wm1, bm1, Wm2, bm2, Wm3, bm3, out);
}
